// MultiHeadAttentionBlock_76209899700433
// MI455X (gfx1250) — matrix operand order repaired and checked by reading, not compiled, not run on hardware
//
#include <hip/hip_runtime.h>

// ---------------------------------------------------------------------------
// Transformer encoder block (MHA + FFN + 2x LayerNorm) for gfx1250 (MI455X).
// All GEMMs run on V_WMMA_F32_16X16X32_BF16 (wave32 WMMA, f32 accumulate).
// Each wave owns a 64x64 C block (4x4 WMMA tiles): 16 b128 loads per 16 WMMAs.
// ---------------------------------------------------------------------------

#define EMBED 1024
#define NHEAD 16
#define HDIM  64
#define FFDIM 4096
#define BB    4
#define LL    2048
#define BL    (BB * LL)          // 8192 rows
#define EPS   1e-5f

typedef __bf16 bf16;
typedef __attribute__((ext_vector_type(16))) __bf16 v16bf;
typedef __attribute__((ext_vector_type(8)))  __bf16 v8bf;
typedef __attribute__((ext_vector_type(8)))  float  v8f;

// ---- bf16 helpers (round-to-nearest-even) ---------------------------------
__device__ __forceinline__ bf16 f2bf(float f) {
    unsigned u = __builtin_bit_cast(unsigned, f);
    u += 0x7FFFu + ((u >> 16) & 1u);
    unsigned short h = (unsigned short)(u >> 16);
    return __builtin_bit_cast(bf16, h);
}

// ---- WMMA fragment loaders (layouts per CDNA5 ISA 7.12.2, wave32) ---------
// A: 16x32 bf16 (MxK). row M = lane&15; half=lane>>4; v0..3 hold K=half*8+0..7,
// v4..7 hold K=16+half*8+0..7. Two 16-byte loads per lane.
__device__ __forceinline__ v16bf load_a_bf16(const bf16* A, int lda,
                                             int m0, int k0, int lane) {
    int row  = m0 + (lane & 15);
    int half = lane >> 4;
    const bf16* p = A + (size_t)row * lda + k0 + half * 8;
    v8bf lo = *(const v8bf*)(p);
    v8bf hi = *(const v8bf*)(p + 16);
    return __builtin_shufflevector(lo, hi, 0,1,2,3,4,5,6,7,8,9,10,11,12,13,14,15);
}

// B: 32x16 bf16 (KxN), supplied as weight matrix W[N][K] (torch layout).
// col N = lane&15; half=lane>>4; v0..3 hold K=half*8+0..7, v4..7 hold K=16+half*8+0..7.
__device__ __forceinline__ v16bf load_b_bf16(const bf16* B, int ldb,
                                             int n0, int k0, int lane) {
    int col  = n0 + (lane & 15);
    int half = lane >> 4;
    const bf16* p = B + (size_t)col * ldb + k0 + half * 8;
    v8bf lo = *(const v8bf*)(p);
    v8bf hi = *(const v8bf*)(p + 16);
    return __builtin_shufflevector(lo, hi, 0,1,2,3,4,5,6,7,8,9,10,11,12,13,14,15);
}

// A fragment from an f32 matrix (attention probabilities), converted in-reg.
__device__ __forceinline__ v16bf load_a_f32(const float* A, int lda,
                                            int m0, int k0, int lane) {
    int row  = m0 + (lane & 15);
    int half = lane >> 4;
    const float* p = A + (size_t)row * lda + k0 + half * 8;
    v16bf a = {};
#pragma unroll
    for (int i = 0; i < 8; ++i) a[i] = f2bf(p[i]);
#pragma unroll
    for (int i = 0; i < 8; ++i) a[8 + i] = f2bf(p[16 + i]);
    return a;
}

__device__ __forceinline__ v8f wmma_bf16(v16bf a, v16bf b, v8f c) {
    return __builtin_amdgcn_wmma_f32_16x16x32_bf16(
        /*neg_a=*/false, a, /*neg_b=*/false, b,
        /*c_mod=*/(short)0, c, /*reuse_a=*/false, /*reuse_b=*/false);
}

// ---------------------------------------------------------------------------
// Generic GEMM: C[M,N] = A[M,K] @ B[N,K]^T (+bias)(+res)(relu), f32 or bf16 out
// block = (32,4): 4 waves; each wave owns a 64x64 C block; block covers 128x128
// ---------------------------------------------------------------------------
template <bool BIAS, bool RELU, bool RES, bool OUTBF16>
__global__ __launch_bounds__(128)
void gemm_bf16_kernel(const bf16* __restrict__ A, const bf16* __restrict__ B,
                      const float* __restrict__ bias, const float* __restrict__ res,
                      float* __restrict__ Cf, bf16* __restrict__ Cb,
                      int K, int lda, int ldb, int ldc) {
    const int lane = threadIdx.x;
    const int w    = threadIdx.y;
    const int n0   = blockIdx.x * 128 + (w & 1) * 64;
    const int m0   = blockIdx.y * 128 + (w >> 1) * 64;

    v8f c[4][4] = {};
    for (int k0 = 0; k0 < K; k0 += 32) {
        v16bf a[4], b[4];
#pragma unroll
        for (int i = 0; i < 4; ++i) a[i] = load_a_bf16(A, lda, m0 + 16 * i, k0, lane);
#pragma unroll
        for (int j = 0; j < 4; ++j) b[j] = load_b_bf16(B, ldb, n0 + 16 * j, k0, lane);
#pragma unroll
        for (int i = 0; i < 4; ++i)
#pragma unroll
            for (int j = 0; j < 4; ++j)
                c[i][j] = wmma_bf16(a[i], b[j], c[i][j]);
    }

    const int half = lane >> 4, nn = lane & 15;
#pragma unroll
    for (int mi = 0; mi < 4; ++mi)
#pragma unroll
        for (int ni = 0; ni < 4; ++ni)
#pragma unroll
            for (int v = 0; v < 8; ++v) {
                int row = m0 + mi * 16 + half * 8 + v;
                int col = n0 + ni * 16 + nn;
                float x = c[mi][ni][v];
                if (BIAS) x += bias[col];
                if (RES)  x += res[(size_t)row * ldc + col];
                if (RELU) x = fmaxf(x, 0.0f);
                if (OUTBF16) Cb[(size_t)row * ldc + col] = f2bf(x);
                else         Cf[(size_t)row * ldc + col] = x;
            }
}

// ---------------------------------------------------------------------------
// scores[b,h] = (Q[b,h] @ K[b,h]^T) * scale ; Q/K rows strided inside x-proj
// grid (L/128, L/128, B*H)
// ---------------------------------------------------------------------------
__global__ __launch_bounds__(128)
void scores_kernel(const bf16* __restrict__ Q, const bf16* __restrict__ Km,
                   float* __restrict__ attn, float scale) {
    const int lane = threadIdx.x;
    const int w    = threadIdx.y;
    const int z    = blockIdx.z;
    const int b    = z / NHEAD, h = z % NHEAD;
    const int n0   = blockIdx.x * 128 + (w & 1) * 64;
    const int m0   = blockIdx.y * 128 + (w >> 1) * 64;

    const bf16* Ab = Q  + (size_t)b * LL * EMBED + h * HDIM;
    const bf16* Bb = Km + (size_t)b * LL * EMBED + h * HDIM;
    float*      Cc = attn + (size_t)z * LL * LL;

    v8f c[4][4] = {};
    for (int k0 = 0; k0 < HDIM; k0 += 32) {
        v16bf a[4], b2[4];
#pragma unroll
        for (int i = 0; i < 4; ++i) a[i]  = load_a_bf16(Ab, EMBED, m0 + 16 * i, k0, lane);
#pragma unroll
        for (int j = 0; j < 4; ++j) b2[j] = load_b_bf16(Bb, EMBED, n0 + 16 * j, k0, lane);
#pragma unroll
        for (int i = 0; i < 4; ++i)
#pragma unroll
            for (int j = 0; j < 4; ++j)
                c[i][j] = wmma_bf16(a[i], b2[j], c[i][j]);
    }

    const int half = lane >> 4, nn = lane & 15;
#pragma unroll
    for (int mi = 0; mi < 4; ++mi)
#pragma unroll
        for (int ni = 0; ni < 4; ++ni)
#pragma unroll
            for (int v = 0; v < 8; ++v) {
                int row = m0 + mi * 16 + half * 8 + v;
                int col = n0 + ni * 16 + nn;
                Cc[(size_t)row * LL + col] = c[mi][ni][v] * scale;
            }
}

// ---------------------------------------------------------------------------
// out_h = attn @ V  (A is f32 -> bf16 in regs; V pre-transposed to [B,H,D,L])
// one wave covers full Dh=64 in N and 64 rows in M; block covers 256 rows.
// grid (1, L/256, B*H)
// ---------------------------------------------------------------------------
__global__ __launch_bounds__(128)
void av_kernel(const float* __restrict__ attn, const bf16* __restrict__ Vt,
               bf16* __restrict__ concat) {
    const int lane = threadIdx.x;
    const int w    = threadIdx.y;
    const int z    = blockIdx.z;
    const int b    = z / NHEAD, h = z % NHEAD;
    const int n0   = 0;                                   // full Dh per wave
    const int m0   = blockIdx.y * 256 + w * 64;           // l

    const float* Ab = attn + (size_t)z * LL * LL;
    const bf16*  Bb = Vt   + (size_t)z * HDIM * LL;       // [Dh][L], ldb = L
    bf16*        Cc = concat + (size_t)b * LL * EMBED + h * HDIM;

    v8f c[4][4] = {};
    for (int k0 = 0; k0 < LL; k0 += 32) {
        v16bf a[4], b2[4];
#pragma unroll
        for (int i = 0; i < 4; ++i) a[i]  = load_a_f32(Ab, LL, m0 + 16 * i, k0, lane);
#pragma unroll
        for (int j = 0; j < 4; ++j) b2[j] = load_b_bf16(Bb, LL, n0 + 16 * j, k0, lane);
#pragma unroll
        for (int i = 0; i < 4; ++i)
#pragma unroll
            for (int j = 0; j < 4; ++j)
                c[i][j] = wmma_bf16(a[i], b2[j], c[i][j]);
    }

    const int half = lane >> 4, nn = lane & 15;
#pragma unroll
    for (int mi = 0; mi < 4; ++mi)
#pragma unroll
        for (int ni = 0; ni < 4; ++ni)
#pragma unroll
            for (int v = 0; v < 8; ++v) {
                int row = m0 + mi * 16 + half * 8 + v;
                int col = n0 + ni * 16 + nn;
                Cc[(size_t)row * EMBED + col] = f2bf(c[mi][ni][v]);
            }
}

// ---------------------------------------------------------------------------
// Elementwise / reduction kernels
// ---------------------------------------------------------------------------
__global__ void f32_to_bf16_kernel(const float* __restrict__ in,
                                   bf16* __restrict__ out, size_t n) {
    size_t i = (size_t)blockIdx.x * blockDim.x + threadIdx.x;
    size_t stride = (size_t)gridDim.x * blockDim.x;
    for (; i < n; i += stride) out[i] = f2bf(in[i]);
}

// V[B,L,H*D] bf16 -> Vt[B,H,D,L] bf16
__global__ void transpose_v_kernel(const bf16* __restrict__ V,
                                   bf16* __restrict__ Vt) {
    size_t i = (size_t)blockIdx.x * blockDim.x + threadIdx.x;
    if (i >= (size_t)BL * EMBED) return;
    int c = (int)(i % EMBED);
    size_t r = i / EMBED;          // b*L + l
    int b = (int)(r / LL), l = (int)(r % LL);
    int h = c / HDIM, d = c % HDIM;
    Vt[(((size_t)(b * NHEAD + h)) * HDIM + d) * LL + l] = V[i];
}

// in-place row softmax over 2048-wide rows
__global__ __launch_bounds__(256)
void softmax_rows_kernel(float* __restrict__ attn) {
    __shared__ float red[256];
    const int tid = threadIdx.x;
    float* p = attn + (size_t)blockIdx.x * LL;

    float m = -3.0e38f;
#pragma unroll
    for (int j = 0; j < LL / 256; ++j) m = fmaxf(m, p[tid + j * 256]);
    red[tid] = m;
    __syncthreads();
    for (int s = 128; s > 0; s >>= 1) {
        if (tid < s) red[tid] = fmaxf(red[tid], red[tid + s]);
        __syncthreads();
    }
    m = red[0];
    __syncthreads();

    float sum = 0.0f;
    float e[LL / 256];
#pragma unroll
    for (int j = 0; j < LL / 256; ++j) {
        e[j] = expf(p[tid + j * 256] - m);
        sum += e[j];
    }
    red[tid] = sum;
    __syncthreads();
    for (int s = 128; s > 0; s >>= 1) {
        if (tid < s) red[tid] += red[tid + s];
        __syncthreads();
    }
    float inv = 1.0f / red[0];
#pragma unroll
    for (int j = 0; j < LL / 256; ++j) p[tid + j * 256] = e[j] * inv;
}

// layernorm over 1024-wide rows; optionally also emits a bf16 copy
template <bool WRITE_BF16>
__global__ __launch_bounds__(256)
void layernorm_kernel(const float* __restrict__ in, const float* __restrict__ g,
                      const float* __restrict__ bta, float* __restrict__ out,
                      bf16* __restrict__ out_bf) {
    __shared__ float rs[256], rq[256];
    const int tid = threadIdx.x;
    const float* p = in + (size_t)blockIdx.x * EMBED;

    float s = 0.0f, q = 0.0f;
    float x[EMBED / 256];
#pragma unroll
    for (int j = 0; j < EMBED / 256; ++j) {
        x[j] = p[tid + j * 256];
        s += x[j];
        q += x[j] * x[j];
    }
    rs[tid] = s; rq[tid] = q;
    __syncthreads();
    for (int st = 128; st > 0; st >>= 1) {
        if (tid < st) { rs[tid] += rs[tid + st]; rq[tid] += rq[tid + st]; }
        __syncthreads();
    }
    const float mu  = rs[0] * (1.0f / EMBED);
    const float var = rq[0] * (1.0f / EMBED) - mu * mu;
    const float rst = rsqrtf(var + EPS);

#pragma unroll
    for (int j = 0; j < EMBED / 256; ++j) {
        int col = tid + j * 256;
        float y = (x[j] - mu) * rst * g[col] + bta[col];
        out[(size_t)blockIdx.x * EMBED + col] = y;
        if (WRITE_BF16) out_bf[(size_t)blockIdx.x * EMBED + col] = f2bf(y);
    }
}

// ---------------------------------------------------------------------------
// Host launcher
// ---------------------------------------------------------------------------
extern "C" void kernel_launch(void* const* d_in, const int* in_sizes, int n_in,
                              void* d_out, int out_size, void* d_ws, size_t ws_size,
                              hipStream_t stream) {
    const float* x   = (const float*)d_in[0];
    const float* Wq  = (const float*)d_in[1];
    const float* Wk  = (const float*)d_in[2];
    const float* Wv  = (const float*)d_in[3];
    const float* Wo  = (const float*)d_in[4];
    const float* bo  = (const float*)d_in[5];
    const float* g1  = (const float*)d_in[6];
    const float* b1  = (const float*)d_in[7];
    const float* W1  = (const float*)d_in[8];
    const float* bf1 = (const float*)d_in[9];
    const float* W2  = (const float*)d_in[10];
    const float* bf2 = (const float*)d_in[11];
    const float* g2  = (const float*)d_in[12];
    const float* b2  = (const float*)d_in[13];

    float* out_f = (float*)d_out;                         // [B,L,E]
    float* attn  = out_f + (size_t)BL * EMBED;            // [B,H,L,L]

    // workspace bump allocator
    char* w = (char*)d_ws;
    auto alloc = [&](size_t bytes) {
        void* p = (void*)w;
        w += (bytes + 255) & ~(size_t)255;
        return p;
    };
    bf16* x_bf     = (bf16*)alloc((size_t)BL * EMBED * 2);
    bf16* Wq_bf    = (bf16*)alloc((size_t)EMBED * EMBED * 2);
    bf16* Wk_bf    = (bf16*)alloc((size_t)EMBED * EMBED * 2);
    bf16* Wv_bf    = (bf16*)alloc((size_t)EMBED * EMBED * 2);
    bf16* Wo_bf    = (bf16*)alloc((size_t)EMBED * EMBED * 2);
    bf16* W1_bf    = (bf16*)alloc((size_t)FFDIM * EMBED * 2);
    bf16* W2_bf    = (bf16*)alloc((size_t)EMBED * FFDIM * 2);
    bf16* Q_bf     = (bf16*)alloc((size_t)BL * EMBED * 2);
    bf16* K_bf     = (bf16*)alloc((size_t)BL * EMBED * 2);
    bf16* V_bf     = (bf16*)alloc((size_t)BL * EMBED * 2);
    bf16* Vt_bf    = (bf16*)alloc((size_t)BL * EMBED * 2);
    bf16* concat   = (bf16*)alloc((size_t)BL * EMBED * 2);
    float* h_pre   = (float*)alloc((size_t)BL * EMBED * 4);
    float* h_f32   = (float*)alloc((size_t)BL * EMBED * 4);
    bf16* h_bf     = (bf16*)alloc((size_t)BL * EMBED * 2);
    bf16* ff1_bf   = (bf16*)alloc((size_t)BL * FFDIM * 2);
    float* h2_pre  = (float*)alloc((size_t)BL * EMBED * 4);

    // 1. fp32 -> bf16 conversions
    auto cvt = [&](const float* src, bf16* dst, size_t n) {
        int blocks = (int)((n + 255) / 256);
        if (blocks > 4096) blocks = 4096;
        f32_to_bf16_kernel<<<blocks, 256, 0, stream>>>(src, dst, n);
    };
    cvt(x,  x_bf,  (size_t)BL * EMBED);
    cvt(Wq, Wq_bf, (size_t)EMBED * EMBED);
    cvt(Wk, Wk_bf, (size_t)EMBED * EMBED);
    cvt(Wv, Wv_bf, (size_t)EMBED * EMBED);
    cvt(Wo, Wo_bf, (size_t)EMBED * EMBED);
    cvt(W1, W1_bf, (size_t)FFDIM * EMBED);
    cvt(W2, W2_bf, (size_t)EMBED * FFDIM);

    dim3 blk(32, 4);

    // 2. QKV projections: [BL,E] @ [E,E]^T -> bf16
    {
        dim3 grd(EMBED / 128, BL / 128);
        gemm_bf16_kernel<false, false, false, true><<<grd, blk, 0, stream>>>(
            x_bf, Wq_bf, nullptr, nullptr, nullptr, Q_bf, EMBED, EMBED, EMBED, EMBED);
        gemm_bf16_kernel<false, false, false, true><<<grd, blk, 0, stream>>>(
            x_bf, Wk_bf, nullptr, nullptr, nullptr, K_bf, EMBED, EMBED, EMBED, EMBED);
        gemm_bf16_kernel<false, false, false, true><<<grd, blk, 0, stream>>>(
            x_bf, Wv_bf, nullptr, nullptr, nullptr, V_bf, EMBED, EMBED, EMBED, EMBED);
    }

    // 3. V -> Vt [B,H,D,L]
    {
        size_t n = (size_t)BL * EMBED;
        transpose_v_kernel<<<(int)((n + 255) / 256), 256, 0, stream>>>(V_bf, Vt_bf);
    }

    // 4. scores = QK^T / sqrt(Dh) -> attn (raw)
    {
        dim3 grd(LL / 128, LL / 128, BB * NHEAD);
        scores_kernel<<<grd, blk, 0, stream>>>(Q_bf, K_bf, attn, 0.125f);
    }

    // 5. softmax rows (in place on output attn)
    softmax_rows_kernel<<<BB * NHEAD * LL, 256, 0, stream>>>(attn);

    // 6. out_h = attn @ V -> concat (bf16)
    {
        dim3 grd(1, LL / 256, BB * NHEAD);
        av_kernel<<<grd, blk, 0, stream>>>(attn, Vt_bf, concat);
    }

    // 7. mha_out = concat @ Wo^T + bo; h_pre = x + mha_out
    {
        dim3 grd(EMBED / 128, BL / 128);
        gemm_bf16_kernel<true, false, true, false><<<grd, blk, 0, stream>>>(
            concat, Wo_bf, bo, x, h_pre, nullptr, EMBED, EMBED, EMBED, EMBED);
    }

    // 8. h = LN1(h_pre) (f32 + bf16 copy)
    layernorm_kernel<true><<<BL, 256, 0, stream>>>(h_pre, g1, b1, h_f32, h_bf);

    // 9. ff1 = relu(h @ W1^T + bf1) -> bf16
    {
        dim3 grd(FFDIM / 128, BL / 128);
        gemm_bf16_kernel<true, true, false, true><<<grd, blk, 0, stream>>>(
            h_bf, W1_bf, bf1, nullptr, nullptr, ff1_bf, EMBED, EMBED, EMBED, FFDIM);
    }

    // 10. h2_pre = h + (ff1 @ W2^T + bf2)
    {
        dim3 grd(EMBED / 128, BL / 128);
        gemm_bf16_kernel<true, false, true, false><<<grd, blk, 0, stream>>>(
            ff1_bf, W2_bf, bf2, h_f32, h2_pre, nullptr, FFDIM, FFDIM, FFDIM, EMBED);
    }

    // 11. out = LN2(h2_pre)
    layernorm_kernel<false><<<BL, 256, 0, stream>>>(h2_pre, g2, b2, out_f, nullptr);

    (void)in_sizes; (void)n_in; (void)out_size; (void)ws_size;
}
